// SelfAttention_39444979646491
// MI455X (gfx1250) — matrix operand order repaired and checked by reading, not compiled, not run on hardware
//
#include <hip/hip_runtime.h>
#include <hip/hip_bf16.h>

typedef __attribute__((ext_vector_type(16))) __bf16 v16bf;
typedef __attribute__((ext_vector_type(8)))  float  v8f;
typedef unsigned int u32x4 __attribute__((ext_vector_type(4)));
typedef int          i32x4 __attribute__((ext_vector_type(4)));
typedef int          i32x8 __attribute__((ext_vector_type(8)));

constexpr int Bc = 4, Lc = 2048, Cc = 1024, Hc = 16, Dc = 64;
constexpr float MAX_SCALE = 4.6051702f; // ln(100)

__device__ __forceinline__ v8f wmma_bf16(v16bf a, v16bf b, v8f c) {
  return __builtin_amdgcn_wmma_f32_16x16x32_bf16(false, a, false, b,
                                                 (short)0, c, false, false);
}

// --- Tensor Data Mover support (probe-verified builtin; arity differs by clang) ---
#if defined(__has_builtin)
#if __has_builtin(__builtin_amdgcn_tensor_load_to_lds) && \
    __has_builtin(__builtin_amdgcn_s_wait_tensorcnt)
#define USE_TDM 1
#endif
#endif
#ifndef USE_TDM
#define USE_TDM 0
#endif

#if USE_TDM
// DMA a contiguous [rows x cols] bf16 tile (row stride = row_stride elems) to LDS.
// D# built per CDNA5 ISA section 8.3/8.4 (group0: count/lds/global/type,
// group1: data_size=2B, tensor dims, tile dims, dim0 stride).
__device__ __forceinline__ void tdm_load_2d_bf16(const __bf16* gsrc,
                                                 unsigned lds_addr,
                                                 unsigned rows, unsigned cols,
                                                 unsigned row_stride) {
  unsigned long long ga = (unsigned long long)(size_t)gsrc;
  u32x4 g0;
  g0[0] = 1u;                                            // count=1, user D#
  g0[1] = lds_addr;                                      // LDS byte address
  g0[2] = (unsigned)(ga & 0xFFFFFFFFull);                // global_addr[31:0]
  g0[3] = (unsigned)((ga >> 32) & 0x01FFFFFFull) | (2u << 30); // addr[56:32]|type=2
  unsigned long long td0 = cols, td1 = rows;
  unsigned long long t0 = cols, t1 = rows;
  unsigned long long s0 = row_stride;
  union { unsigned long long q[4]; i32x8 v; } g1;
  g1.q[0] = (1ull << 16)                       // data_size=1 -> 2 bytes
          | ((td0 & 0xFFFFull) << 48);         // tensor_dim0[15:0]
  g1.q[1] = ((td0 >> 16) & 0xFFFFull)          // tensor_dim0[31:16]
          | ((td1 & 0xFFFFFFFFull) << 16)      // tensor_dim1
          | (t0 << 48);                        // tile_dim0
  g1.q[2] = t1                                 // tile_dim1
          | ((s0 & 0xFFFFFFFFull) << 32);      // tensor_dim0_stride[31:0]
  g1.q[3] = ((s0 >> 32) & 0xFFFFull);          // stride[47:32], dim1_stride=0
  i32x4 z4 = {0, 0, 0, 0};
#if __clang_major__ >= 23
  i32x8 z8 = {0, 0, 0, 0, 0, 0, 0, 0};
  __builtin_amdgcn_tensor_load_to_lds(g0, g1.v, z4, z4, z8, 0);
#else
  __builtin_amdgcn_tensor_load_to_lds(g0, g1.v, z4, z4, 0);
#endif
}
#endif

// ---------------------------------------------------------------------------
// Kernel 1: QKV = x @ W^T + bias -> q,k,v bf16 in [B,H,L,D]
// M=8192, N=3072, K=1024. Block tile 128x64, 8 waves, 32x32 (4 WMMA) per wave.
// ---------------------------------------------------------------------------
__global__ __launch_bounds__(256) void qkv_gemm(
    const float* __restrict__ x, const float* __restrict__ w,
    const float* __restrict__ qbias, const float* __restrict__ vbias,
    __bf16* __restrict__ qO, __bf16* __restrict__ kO, __bf16* __restrict__ vO)
{
  __shared__ __bf16 As[128][36];
  __shared__ __bf16 Bs[64][36];
  const int tid  = threadIdx.x;
  const int lane = tid & 31;
  const int wave = tid >> 5;
  const int wm = wave & 3, wn = wave >> 2;        // 4x2 wave grid
  const int row0 = blockIdx.x * 128;
  const int col0 = blockIdx.y * 64;
  const int half = lane >> 4, l16 = lane & 15, abase = half * 8;
  const int ar = tid >> 3;                        // 0..31 row per pass
  const int ac = (tid & 7) * 4;                   // col (float4 granularity)

  v8f acc[2][2] = {};

  for (int k0 = 0; k0 < Cc; k0 += 32) {
#pragma unroll
    for (int p = 0; p < 4; ++p) {                 // A tile 128x32 f32->bf16
      int r = ar + p * 32;
      float4 f = *(const float4*)&x[(size_t)(row0 + r) * Cc + k0 + ac];
      As[r][ac + 0] = (__bf16)f.x; As[r][ac + 1] = (__bf16)f.y;
      As[r][ac + 2] = (__bf16)f.z; As[r][ac + 3] = (__bf16)f.w;
    }
#pragma unroll
    for (int p = 0; p < 2; ++p) {                 // B tile 64x32 f32->bf16
      int r = ar + p * 32;
      float4 f = *(const float4*)&w[(size_t)(col0 + r) * Cc + k0 + ac];
      Bs[r][ac + 0] = (__bf16)f.x; Bs[r][ac + 1] = (__bf16)f.y;
      Bs[r][ac + 2] = (__bf16)f.z; Bs[r][ac + 3] = (__bf16)f.w;
    }
    if (k0 + 32 < Cc)                             // gfx1250 global_prefetch_b8
      __builtin_prefetch(&x[(size_t)(row0 + ar) * Cc + k0 + 32 + ac], 0, 1);
    __syncthreads();

    v16bf a[2], b[2];
#pragma unroll
    for (int e = 0; e < 16; ++e) {
      int kfA = half * 16 + e;
      a[0][e] = As[wm * 32 + l16][kfA];
      a[1][e] = As[wm * 32 + 16 + l16][kfA];
      int kfB = half ? (16 + e) : e;
      b[0][e] = Bs[wn * 32 + l16][kfB];
      b[1][e] = Bs[wn * 32 + 16 + l16][kfB];
    }
#pragma unroll
    for (int i = 0; i < 2; ++i)
#pragma unroll
      for (int j = 0; j < 2; ++j)
        acc[i][j] = wmma_bf16(a[i], b[j], acc[i][j]);
    __syncthreads();
  }

#pragma unroll
  for (int j = 0; j < 2; ++j) {
    const int n   = col0 + wn * 32 + j * 16 + l16;
    const int mat = n >> 10;                      // 0=q 1=k 2=v
    const int c   = n & (Cc - 1);
    const int h   = c >> 6, d = c & 63;
    const float bias = (mat == 0) ? qbias[c] : (mat == 2 ? vbias[c] : 0.f);
    __bf16* dst = (mat == 0) ? qO : (mat == 1 ? kO : vO);
#pragma unroll
    for (int i = 0; i < 2; ++i)
#pragma unroll
      for (int r = 0; r < 8; ++r) {
        int m   = row0 + wm * 32 + i * 16 + r + 8 * half;
        int bb  = m >> 11;
        int pos = m & (Lc - 1);
        dst[(((size_t)bb * Hc + h) * Lc + pos) * Dc + d] =
            (__bf16)(acc[i][j][r] + bias);
      }
  }
}

// ---------------------------------------------------------------------------
// Kernel 2: in-place L2-norm (+scale for q) + interleaved RoPE on q and k
// ---------------------------------------------------------------------------
__global__ __launch_bounds__(256) void norm_rope(
    __bf16* __restrict__ q, __bf16* __restrict__ k,
    const float* __restrict__ rope, const float* __restrict__ scale_mul)
{
  const int row = blockIdx.x * 256 + threadIdx.x;   // (b*H + h)*L + pos
  const int b   = row >> 15;
  const int rem = row & (Hc * Lc - 1);
  const int h   = rem >> 11;
  const int pos = rem & (Lc - 1);
  const float* cosp = rope + ((size_t)b * 2)     * Lc * 32 + (size_t)pos * 32;
  const float* sinp = rope + ((size_t)b * 2 + 1) * Lc * 32 + (size_t)pos * 32;
  const float sm = __expf(fminf(scale_mul[h], MAX_SCALE));

  float t[64];
#pragma unroll 2
  for (int pass = 0; pass < 2; ++pass) {
    __bf16* rowp = (pass == 0 ? q : k) + (size_t)row * Dc;
    float ss = 0.f;
#pragma unroll
    for (int i = 0; i < 64; ++i) { t[i] = (float)rowp[i]; ss += t[i] * t[i]; }
    float inv = (pass == 0 ? sm : 1.f) / fmaxf(sqrtf(ss), 1e-12f);
#pragma unroll
    for (int i = 0; i < 32; ++i) {
      float cc = cosp[i], s = sinp[i];
      float t0 = t[2 * i] * inv, t1 = t[2 * i + 1] * inv;
      rowp[2 * i]     = (__bf16)(t0 * cc - t1 * s);
      rowp[2 * i + 1] = (__bf16)(t0 * s + t1 * cc);
    }
  }
}

// ---------------------------------------------------------------------------
// Kernel 3: causal flash attention. One wave per 16-query tile per (b,h).
// K/V chunks (32x64 bf16) staged to LDS via TDM tensor_load_to_lds when
// available (s_wait_tensorcnt sync), else cooperative b128 copies.
// ---------------------------------------------------------------------------
__global__ __launch_bounds__(32) void flash_attn(
    const __bf16* __restrict__ Q, const __bf16* __restrict__ K,
    const __bf16* __restrict__ V, __bf16* __restrict__ O)
{
  __shared__ __bf16 Ks[32][64];     // contiguous: TDM writes linear tiles
  __shared__ __bf16 Vs[32][64];
  __shared__ float  Ps[16][33];
  const int lane = threadIdx.x;
  const int half = lane >> 4, l16 = lane & 15;
  const int bh = blockIdx.y;
  const int b = bh >> 4, h = bh & 15;
  const int q0 = blockIdx.x * 16;
  const __bf16* qrow = Q + (size_t)bh * Lc * Dc;
  const __bf16* krow = K + (size_t)bh * Lc * Dc;
  const __bf16* vrow = V + (size_t)bh * Lc * Dc;
  const int abase = half * 8;

  v16bf qa0, qa1;
#pragma unroll
  for (int e = 0; e < 16; ++e) {
    int kf = half * 16 + e;
    qa0[e] = qrow[(size_t)(q0 + l16) * Dc + kf];
    qa1[e] = qrow[(size_t)(q0 + l16) * Dc + 32 + kf];
  }

  v8f oacc[4] = {};
  float rowmax[8], rowsum[8];
#pragma unroll
  for (int r = 0; r < 8; ++r) { rowmax[r] = -3.0e38f; rowsum[r] = 0.f; }

  for (int kb = 0; kb <= q0 + 15; kb += 32) {
    // ---- stage K and V chunks into LDS ----
#if USE_TDM
    tdm_load_2d_bf16(krow + (size_t)kb * Dc, (unsigned)(size_t)&Ks[0][0],
                     32u, 64u, 64u);
    tdm_load_2d_bf16(vrow + (size_t)kb * Dc, (unsigned)(size_t)&Vs[0][0],
                     32u, 64u, 64u);
    __builtin_amdgcn_s_wait_tensorcnt(0);
    __syncthreads();
#else
    {
      const uint4* gk = (const uint4*)(krow + (size_t)kb * Dc);
      const uint4* gv = (const uint4*)(vrow + (size_t)kb * Dc);
      uint4* lk = (uint4*)&Ks[0][0];
      uint4* lv = (uint4*)&Vs[0][0];
#pragma unroll
      for (int j = 0; j < 8; ++j) {
        lk[lane + 32 * j] = gk[lane + 32 * j];
        lv[lane + 32 * j] = gv[lane + 32 * j];
      }
    }
    __syncthreads();
#endif

    // ---- S = Q K^T : 16x32 scores via 4 WMMAs ----
    v8f sacc[2] = {};
#pragma unroll
    for (int nt = 0; nt < 2; ++nt) {
      const int kl = nt * 16 + l16;                 // key within chunk
#pragma unroll
      for (int dh = 0; dh < 2; ++dh) {
        v16bf bf;
#pragma unroll
        for (int e = 0; e < 16; ++e) {
          int kf = half ? (16 + e) : e;
          bf[e] = Ks[kl][dh * 32 + kf];             // contiguous -> ds_load_b128
        }
        sacc[nt] = wmma_bf16(dh ? qa1 : qa0, bf, sacc[nt]);
      }
    }

    // ---- causal mask + online softmax ----
    float pv0[8], pv1[8];
#pragma unroll
    for (int r = 0; r < 8; ++r) {
      const int qidx = q0 + r + 8 * half;
      float s0 = sacc[0][r]; if (kb + l16 > qidx)      s0 = -3.0e38f;
      float s1 = sacc[1][r]; if (kb + 16 + l16 > qidx) s1 = -3.0e38f;
      float mx = fmaxf(s0, s1);
#pragma unroll
      for (int off = 1; off < 16; off <<= 1) mx = fmaxf(mx, __shfl_xor(mx, off, 32));
      const float nmax = fmaxf(rowmax[r], mx);
      const float alpha = __expf(rowmax[r] - nmax);
      rowmax[r] = nmax;
      const float p0 = (s0 <= -1e37f) ? 0.f : __expf(s0 - nmax);
      const float p1 = (s1 <= -1e37f) ? 0.f : __expf(s1 - nmax);
      pv0[r] = p0; pv1[r] = p1;
      float psum = p0 + p1;
#pragma unroll
      for (int off = 1; off < 16; off <<= 1) psum += __shfl_xor(psum, off, 32);
      rowsum[r] = rowsum[r] * alpha + psum;
#pragma unroll
      for (int nc = 0; nc < 4; ++nc) oacc[nc][r] *= alpha;
    }

    // ---- re-pack P (C layout) -> A fragment via LDS ----
#pragma unroll
    for (int r = 0; r < 8; ++r) {
      Ps[r + 8 * half][l16]      = pv0[r];
      Ps[r + 8 * half][16 + l16] = pv1[r];
    }
    __syncthreads();
    v16bf pa;
#pragma unroll
    for (int e = 0; e < 16; ++e) {
      int kf = half * 16 + e;
      pa[e] = (__bf16)Ps[l16][kf];
    }
    __syncthreads();

    // ---- O += P @ V (4 WMMAs over 64 dims) ----
#pragma unroll
    for (int nc = 0; nc < 4; ++nc) {
      v16bf vb;
#pragma unroll
      for (int e = 0; e < 16; ++e) {
        int kf = half ? (16 + e) : e;
        vb[e] = Vs[kf][nc * 16 + l16];
      }
      oacc[nc] = wmma_bf16(pa, vb, oacc[nc]);
    }
    __syncthreads();   // protect Ks/Vs before next chunk's staging
  }

#pragma unroll
  for (int r = 0; r < 8; ++r) {
    const float invs = 1.f / rowsum[r];
    const int m = q0 + r + 8 * half;
#pragma unroll
    for (int nc = 0; nc < 4; ++nc)
      O[((size_t)b * Lc + m) * Cc + h * Dc + nc * 16 + l16] =
          (__bf16)(oacc[nc][r] * invs);
  }
}

// ---------------------------------------------------------------------------
// Kernel 4: Y = O @ proj_w^T + proj_b (M=8192, N=K=1024), f32 out
// ---------------------------------------------------------------------------
__global__ __launch_bounds__(256) void proj_gemm(
    const __bf16* __restrict__ A, const float* __restrict__ w,
    const float* __restrict__ bias, float* __restrict__ Y)
{
  __shared__ __bf16 As[128][36];
  __shared__ __bf16 Bs[64][36];
  const int tid  = threadIdx.x;
  const int lane = tid & 31;
  const int wave = tid >> 5;
  const int wm = wave & 3, wn = wave >> 2;
  const int row0 = blockIdx.x * 128;
  const int col0 = blockIdx.y * 64;
  const int half = lane >> 4, l16 = lane & 15, abase = half * 8;
  const int ar = tid >> 3;
  const int ac = (tid & 7) * 4;
  const int br = tid >> 2;              // 0..63 for bf16 A copy (8 elems)
  const int bc = (tid & 3) * 8;

  v8f acc[2][2] = {};

  for (int k0 = 0; k0 < Cc; k0 += 32) {
#pragma unroll
    for (int p = 0; p < 2; ++p) {       // A tile 128x32 bf16 copy (b128 loads)
      int r = br + p * 64;
      union { uint4 u; __bf16 hx[8]; } t;
      t.u = *(const uint4*)&A[(size_t)(row0 + r) * Cc + k0 + bc];
#pragma unroll
      for (int e = 0; e < 8; ++e) As[r][bc + e] = t.hx[e];
    }
#pragma unroll
    for (int p = 0; p < 2; ++p) {       // B tile 64x32 f32->bf16
      int r = ar + p * 32;
      float4 f = *(const float4*)&w[(size_t)(col0 + r) * Cc + k0 + ac];
      Bs[r][ac + 0] = (__bf16)f.x; Bs[r][ac + 1] = (__bf16)f.y;
      Bs[r][ac + 2] = (__bf16)f.z; Bs[r][ac + 3] = (__bf16)f.w;
    }
    if (k0 + 32 < Cc)
      __builtin_prefetch(&A[(size_t)(row0 + br) * Cc + k0 + 32 + bc], 0, 1);
    __syncthreads();

    v16bf a[2], b[2];
#pragma unroll
    for (int e = 0; e < 16; ++e) {
      int kfA = half * 16 + e;
      a[0][e] = As[wm * 32 + l16][kfA];
      a[1][e] = As[wm * 32 + 16 + l16][kfA];
      int kfB = half ? (16 + e) : e;
      b[0][e] = Bs[wn * 32 + l16][kfB];
      b[1][e] = Bs[wn * 32 + 16 + l16][kfB];
    }
#pragma unroll
    for (int i = 0; i < 2; ++i)
#pragma unroll
      for (int j = 0; j < 2; ++j)
        acc[i][j] = wmma_bf16(a[i], b[j], acc[i][j]);
    __syncthreads();
  }

#pragma unroll
  for (int j = 0; j < 2; ++j) {
    const int n = col0 + wn * 32 + j * 16 + l16;
    const float bs = bias[n];
#pragma unroll
    for (int i = 0; i < 2; ++i)
#pragma unroll
      for (int r = 0; r < 8; ++r) {
        int m = row0 + wm * 32 + i * 16 + r + 8 * half;
        Y[(size_t)m * Cc + n] = acc[i][j][r] + bs;
      }
  }
}

// ---------------------------------------------------------------------------
extern "C" void kernel_launch(void* const* d_in, const int* in_sizes, int n_in,
                              void* d_out, int out_size, void* d_ws, size_t ws_size,
                              hipStream_t stream) {
  const float* x         = (const float*)d_in[0];
  // d_in[1] = attn_bias (unused: causal mask computed analytically)
  const float* rope      = (const float*)d_in[2];
  const float* w_qkv     = (const float*)d_in[3];
  const float* q_bias    = (const float*)d_in[4];
  const float* v_bias    = (const float*)d_in[5];
  const float* proj_w    = (const float*)d_in[6];
  const float* proj_b    = (const float*)d_in[7];
  const float* scale_mul = (const float*)d_in[8];
  float* out = (float*)d_out;

  const size_t SZ = (size_t)Bc * Hc * Lc * Dc;   // 8388608 elements
  __bf16* qB = (__bf16*)d_ws;
  __bf16* kB = qB + SZ;
  __bf16* vB = kB + SZ;
  __bf16* oB = vB + SZ;                           // total 4*SZ*2 = 67.1 MB

  qkv_gemm<<<dim3(8192 / 128, 3072 / 64), 256, 0, stream>>>(
      x, w_qkv, q_bias, v_bias, qB, kB, vB);

  norm_rope<<<dim3((Bc * Hc * Lc) / 256), 256, 0, stream>>>(qB, kB, rope, scale_mul);

  flash_attn<<<dim3(Lc / 16, Bc * Hc), 32, 0, stream>>>(qB, kB, vB, oB);

  proj_gemm<<<dim3(8192 / 128, 1024 / 64), 256, 0, stream>>>(
      oB, proj_w, proj_b, out);
}
